// MambaBlock_63806034150206
// MI455X (gfx1250) — compile-verified, matrix operand order checked by reading, not run on hardware
//
#include <hip/hip_runtime.h>
#include <hip/hip_bf16.h>
#include <math.h>

// Problem constants (match reference)
#define B_ 2
#define L_ 2048
#define D_ 1024
#define N_ 16
#define KC_ 4   // conv kernel width

typedef __bf16 v16bf __attribute__((ext_vector_type(16)));
typedef float  v8f   __attribute__((ext_vector_type(8)));

union Frag32 {
    uint4 q[2];
    v16bf v;
};

__device__ __forceinline__ unsigned short f32_to_bf16(float f) {
    unsigned int u = __float_as_uint(f);
    u += 0x7FFFu + ((u >> 16) & 1u);   // round-to-nearest-even
    return (unsigned short)(u >> 16);
}

// ---------------------------------------------------------------- convert
__global__ void cvt_f32_bf16(const float* __restrict__ in,
                             unsigned short* __restrict__ out, int n) {
    int i = blockIdx.x * blockDim.x + threadIdx.x;
    if (i < n) out[i] = f32_to_bf16(in[i]);
}

// ======================= CDNA5 async global->LDS helpers =================
__device__ __forceinline__ void async_load_b128(unsigned int lds_addr,
                                                const unsigned short* gptr) {
    // GLOBAL_LOAD_ASYNC_TO_LDS_B128: VDST = LDS byte address, VADDR = 64-bit
    // global address. Tracked by ASYNCcnt.
    asm volatile("global_load_async_to_lds_b128 %0, %1, off"
                 :: "v"(lds_addr), "v"((unsigned long long)(uintptr_t)gptr)
                 : "memory");
}
__device__ __forceinline__ void wait_asynccnt0() {
    asm volatile("s_wait_asynccnt 0x0" ::: "memory");
}

// ---------------------------------------------------------------- tiled GEMM
// C[M,Nc] = A[M,Kc](bf16) * Bw[Nc,Kc](bf16)^T (+bias)(+softplus)
// Block tile 128x128, 8 waves (2Mx4N), wave tile 64x32 (8 accumulators).
// K staged through LDS (double-buffered) with async global->LDS B128 loads.
// Requires M%128==0, Nc%128==0, Kc%32==0.
#define LDA_  40                 // halfs per LDS row (32 data + 8 pad = 80B)
#define ABUF_ (128 * LDA_)       // halfs per stage buffer (A or B)

__global__ void __launch_bounds__(256)
gemm_bf16_wmma_tiled(const unsigned short* __restrict__ A,
                     const unsigned short* __restrict__ Bw,
                     const float* __restrict__ bias,
                     float* __restrict__ C,
                     int M, int Nc, int Kc, int mode)
{
    __shared__ __align__(16) unsigned short smem[4 * ABUF_]; // {buf0A,buf0B,buf1A,buf1B}
    const int t     = threadIdx.x;
    const int lane  = t & 31;
    const int wave  = t >> 5;
    const int tilesN = Nc >> 7;
    const int bM = blockIdx.x / tilesN;
    const int bN = blockIdx.x % tilesN;
    const int waveM = wave >> 2;         // 0..1 -> 64 rows each
    const int waveN = wave & 3;          // 0..3 -> 32 cols each
    const int row16 = lane & 15;
    const int kh    = lane >> 4;         // K-half held by this lane

    const unsigned int ldsbase = (unsigned int)(uintptr_t)smem;
    const int rowA0 = bM * 128;
    const int rowB0 = bN * 128;
    const int nk = Kc / 32;

    // stage one 128x32 A tile + 128x32 B tile into LDS buffer `buf`
    auto stage = [&](int kb, int buf) {
        const unsigned int ldsA = ldsbase + (unsigned int)((buf * 2 + 0) * ABUF_) * 2u;
        const unsigned int ldsB = ldsbase + (unsigned int)((buf * 2 + 1) * ABUF_) * 2u;
#pragma unroll
        for (int i = 0; i < 2; ++i) {
            const int c   = t + i * 256;   // chunk id 0..511 (16B chunks)
            const int row = c >> 2;
            const int ko  = (c & 3) * 8;
            async_load_b128(ldsA + (unsigned int)(row * LDA_ + ko) * 2u,
                            A + (size_t)(rowA0 + row) * Kc + kb * 32 + ko);
            async_load_b128(ldsB + (unsigned int)(row * LDA_ + ko) * 2u,
                            Bw + (size_t)(rowB0 + row) * Kc + kb * 32 + ko);
        }
    };

    v8f zero = {};
    v8f acc[4][2];
#pragma unroll
    for (int mt = 0; mt < 4; ++mt)
#pragma unroll
        for (int nt = 0; nt < 2; ++nt) acc[mt][nt] = zero;

    stage(0, 0);

    for (int kb = 0; kb < nk; ++kb) {
        wait_asynccnt0();
        __syncthreads();                         // staged data visible block-wide
        if (kb + 1 < nk) stage(kb + 1, (kb + 1) & 1);

        const unsigned short* sA = smem + ((kb & 1) * 2 + 0) * ABUF_;
        const unsigned short* sB = smem + ((kb & 1) * 2 + 1) * ABUF_;

        Frag32 fa[4];
#pragma unroll
        for (int mt = 0; mt < 4; ++mt) {
            const unsigned short* p = sA + (waveM * 64 + mt * 16 + row16) * LDA_ + kh * 8;
            fa[mt].q[0] = *(const uint4*)(p);        // K = kh*8 .. +7
            fa[mt].q[1] = *(const uint4*)(p + 16);   // K = 16+kh*8 .. +7
        }
#pragma unroll
        for (int nt = 0; nt < 2; ++nt) {
            Frag32 fb;
            const unsigned short* p = sB + (waveN * 32 + nt * 16 + row16) * LDA_ + kh * 8;
            fb.q[0] = *(const uint4*)(p);
            fb.q[1] = *(const uint4*)(p + 16);
#pragma unroll
            for (int mt = 0; mt < 4; ++mt)
                acc[mt][nt] = __builtin_amdgcn_wmma_f32_16x16x32_bf16(
                    false, fa[mt].v, false, fb.v, (short)0, acc[mt][nt], false, false);
        }
        __syncthreads();                         // all waves done reading this buffer
    }

    const int m0 = bM * 128 + waveM * 64;
    const int n0 = bN * 128 + waveN * 32;
#pragma unroll
    for (int nt = 0; nt < 2; ++nt) {
        const int n = n0 + nt * 16 + row16;
        const float bv = (mode >= 1) ? bias[n] : 0.0f;
#pragma unroll
        for (int mt = 0; mt < 4; ++mt) {
#pragma unroll
            for (int v = 0; v < 8; ++v) {
                const int m = m0 + mt * 16 + v + 8 * kh;
                float val = acc[mt][nt][v] + bv;
                if (mode == 2) val = (val > 20.0f) ? val : log1pf(__expf(val));
                C[(size_t)m * Nc + n] = val;
            }
        }
    }
}

// ---------------------------------------------------------------- simple GEMM
// One 16x16 tile per wave (used for the small Nc=32 x-projection GEMM).
__global__ void __launch_bounds__(256)
gemm_bf16_wmma(const unsigned short* __restrict__ A,
               const unsigned short* __restrict__ Bw,
               const float* __restrict__ bias,
               float* __restrict__ C,
               int M, int Nc, int Kc, int mode)
{
    const int lane = threadIdx.x & 31;
    const int wave = threadIdx.x >> 5;
    const int tilesN = Nc >> 4;
    const int tile = blockIdx.x * 8 + wave;
    const int tM = tile / tilesN;
    const int tN = tile % tilesN;

    const int row = lane & 15;
    const int kh  = lane >> 4;

    const unsigned short* Ap = A  + (size_t)(tM * 16 + row) * Kc + kh * 8;
    const unsigned short* Bp = Bw + (size_t)(tN * 16 + row) * Kc + kh * 8;

    v8f acc = {};
    for (int kb = 0; kb < Kc; kb += 32) {
        if (kb + 32 < Kc) {
            __builtin_prefetch(Ap + kb + 32, 0, 1);   // global_prefetch_b8
            __builtin_prefetch(Bp + kb + 32, 0, 1);
        }
        Frag32 fa, fb;
        fa.q[0] = *(const uint4*)(Ap + kb);
        fa.q[1] = *(const uint4*)(Ap + kb + 16);
        fb.q[0] = *(const uint4*)(Bp + kb);
        fb.q[1] = *(const uint4*)(Bp + kb + 16);
        acc = __builtin_amdgcn_wmma_f32_16x16x32_bf16(
                  false, fa.v, false, fb.v, (short)0, acc, false, false);
    }

    const int n = lane & 15;
    float bv = 0.0f;
    if (mode >= 1) bv = bias[tN * 16 + n];

    float* Crow = C + (size_t)(tM * 16) * Nc + tN * 16 + n;
#pragma unroll
    for (int v = 0; v < 8; ++v) {
        const int m = v + 8 * kh;
        float val = acc[v] + bv;
        if (mode == 2) val = (val > 20.0f) ? val : log1pf(__expf(val));
        Crow[(size_t)m * Nc] = val;
    }
}

// ---------------------------------------------------------------- conv+gate
__global__ void __launch_bounds__(256)
conv_gate_kernel(const float* __restrict__ xz,
                 const float* __restrict__ Wc,   // (D,1,KC)
                 const float* __restrict__ bc,   // (D,)
                 float* __restrict__ gate_f,
                 unsigned short* __restrict__ gate_b)
{
    int i = blockIdx.x * blockDim.x + threadIdx.x;
    const int total = B_ * L_ * D_;
    if (i >= total) return;
    const int d = i % D_;
    const int l = (i / D_) % L_;
    const int b = i / (D_ * L_);

    const float* base = xz + (size_t)b * L_ * (2 * D_);
    float s = bc[d];
#pragma unroll
    for (int k = 0; k < KC_; ++k) {
        int ll = l - (KC_ - 1) + k;
        if (ll >= 0) s += base[(size_t)ll * (2 * D_) + d] * Wc[d * KC_ + k];
    }
    const float z = base[(size_t)l * (2 * D_) + D_ + d];
    const float sil = s / (1.0f + __expf(-s));
    const float g = sil * z;
    gate_f[i] = g;
    gate_b[i] = f32_to_bf16(g);
}

// ---------------------------------------------------------------- scan
__global__ void __launch_bounds__(256)
scan_kernel(const float* __restrict__ delta,
            const float* __restrict__ bcmat,   // (B,L,2N): [B_p | C_p]
            const float* __restrict__ gate,
            const float* __restrict__ A_log,   // (D,N)
            const float* __restrict__ x,       // residual
            float* __restrict__ y)
{
    const int lane = threadIdx.x & 31;
    const int wave = blockIdx.x * (blockDim.x >> 5) + (threadIdx.x >> 5);
    const int b  = wave / (D_ / 2);
    const int dp = wave % (D_ / 2);
    const int d  = dp * 2 + (lane >> 4);
    const int n  = lane & 15;

    const float Adn = -__expf(A_log[d * N_ + n]);
    float h = 0.0f;
    for (int l = 0; l < L_; ++l) {
        const size_t off = (size_t)b * L_ + l;
        const float dt = delta[off * D_ + d];
        const float g  = gate [off * D_ + d];
        const float Bv = bcmat[off * (2 * N_) + n];
        const float Cv = bcmat[off * (2 * N_) + N_ + n];
        const float a  = __expf(dt * Adn);
        h = a * h + dt * Bv * g;
        float c = h * Cv;
        c += __shfl_xor(c, 1, 32);
        c += __shfl_xor(c, 2, 32);
        c += __shfl_xor(c, 4, 32);
        c += __shfl_xor(c, 8, 32);
        if (n == 0) y[off * D_ + d] = c + x[off * D_ + d];
    }
}

// ---------------------------------------------------------------- layernorm
__global__ void __launch_bounds__(256)
ln_kernel(const float* __restrict__ y,
          const float* __restrict__ gamma,
          const float* __restrict__ beta,
          unsigned short* __restrict__ out_b)
{
    __shared__ float red[256];
    const int row = blockIdx.x;              // b*L + l
    const float* yr = y + (size_t)row * D_;
    const int t = threadIdx.x;

    float s = 0.0f;
    for (int i = t; i < D_; i += 256) s += yr[i];
    red[t] = s; __syncthreads();
    for (int st = 128; st > 0; st >>= 1) {
        if (t < st) red[t] += red[t + st];
        __syncthreads();
    }
    const float mu = red[0] / D_;
    __syncthreads();

    float v = 0.0f;
    for (int i = t; i < D_; i += 256) { float d = yr[i] - mu; v += d * d; }
    red[t] = v; __syncthreads();
    for (int st = 128; st > 0; st >>= 1) {
        if (t < st) red[t] += red[t + st];
        __syncthreads();
    }
    const float rstd = rsqrtf(red[0] / D_ + 1e-5f);

    for (int i = t; i < D_; i += 256) {
        float o = (yr[i] - mu) * rstd * gamma[i] + beta[i];
        out_b[(size_t)row * D_ + i] = f32_to_bf16(o);
    }
}

// ---------------------------------------------------------------- launcher
extern "C" void kernel_launch(void* const* d_in, const int* in_sizes, int n_in,
                              void* d_out, int out_size, void* d_ws, size_t ws_size,
                              hipStream_t stream) {
    const float* x       = (const float*)d_in[0];   // (B,L,D)
    const float* W_in    = (const float*)d_in[1];   // (2D,D)
    const float* W_conv  = (const float*)d_in[2];   // (D,1,KC)
    const float* b_conv  = (const float*)d_in[3];   // (D,)
    const float* W_xproj = (const float*)d_in[4];   // (2N,D)
    const float* W_dt    = (const float*)d_in[5];   // (D,D)
    const float* b_dt    = (const float*)d_in[6];   // (D,)
    const float* A_log   = (const float*)d_in[7];   // (D,N)
    const float* ln_g    = (const float*)d_in[8];
    const float* ln_b    = (const float*)d_in[9];
    const float* W_out   = (const float*)d_in[10];  // (D,D)
    const float* b_out   = (const float*)d_in[11];  // (D,)
    float* out = (float*)d_out;

    const size_t BLD = (size_t)B_ * L_ * D_;        // 4 Mi elements
    const int    M   = B_ * L_;                     // 4096 rows

    // carve workspace (256B aligned slabs)
    char* base = (char*)d_ws;
    size_t off = 0;
    auto alloc = [&](size_t bytes) -> char* {
        char* p = base + off;
        off = (off + bytes + 255) & ~(size_t)255;
        return p;
    };
    unsigned short* xb     = (unsigned short*)alloc(BLD * 2);
    unsigned short* w_in_b = (unsigned short*)alloc((size_t)2 * D_ * D_ * 2);
    unsigned short* w_dt_b = (unsigned short*)alloc((size_t)D_ * D_ * 2);
    unsigned short* w_xp_b = (unsigned short*)alloc((size_t)2 * N_ * D_ * 2);
    unsigned short* w_ot_b = (unsigned short*)alloc((size_t)D_ * D_ * 2);
    float*          xz     = (float*)alloc(BLD * 2 * 4);
    float*          gate_f = (float*)alloc(BLD * 4);
    unsigned short* gate_b = (unsigned short*)alloc(BLD * 2);
    float*          delta  = (float*)alloc(BLD * 4);
    float*          bcmat  = (float*)alloc((size_t)M * 2 * N_ * 4);
    float*          yf     = (float*)alloc(BLD * 4);
    unsigned short* yb     = (unsigned short*)alloc(BLD * 2);
    (void)ws_size; (void)in_sizes; (void)n_in; (void)out_size;

    // 1) f32 -> bf16 conversions
    auto cvt = [&](const float* src, unsigned short* dst, size_t n) {
        cvt_f32_bf16<<<(unsigned)((n + 255) / 256), 256, 0, stream>>>(src, dst, (int)n);
    };
    cvt(x,       xb,     BLD);
    cvt(W_in,    w_in_b, (size_t)2 * D_ * D_);
    cvt(W_dt,    w_dt_b, (size_t)D_ * D_);
    cvt(W_xproj, w_xp_b, (size_t)2 * N_ * D_);
    cvt(W_out,   w_ot_b, (size_t)D_ * D_);

    auto gemm_big = [&](const unsigned short* A, const unsigned short* Bw,
                        const float* bias, float* C, int Nc, int Kc, int mode) {
        const int blocks = (M / 128) * (Nc / 128);
        gemm_bf16_wmma_tiled<<<blocks, 256, 0, stream>>>(A, Bw, bias, C, M, Nc, Kc, mode);
    };

    // 2) xz = x @ W_in^T                         (4096 x 2048 x 1024)
    gemm_big(xb, w_in_b, nullptr, xz, 2 * D_, D_, 0);

    // 3) depthwise causal conv + silu * z -> gate
    conv_gate_kernel<<<(unsigned)((BLD + 255) / 256), 256, 0, stream>>>(
        xz, W_conv, b_conv, gate_f, gate_b);

    // 4) delta = softplus(gate @ W_dt^T + b_dt)  (4096 x 1024 x 1024)
    gemm_big(gate_b, w_dt_b, b_dt, delta, D_, D_, 2);

    // 5) BC = gate @ W_xproj^T                   (4096 x 32 x 1024) - small N
    {
        const int tiles = (M / 16) * ((2 * N_) / 16);
        gemm_bf16_wmma<<<tiles / 8, 256, 0, stream>>>(
            gate_b, w_xp_b, nullptr, bcmat, M, 2 * N_, D_, 0);
    }

    // 6) selective scan + residual -> yf  (B*D/2 = 1024 waves, 8/block)
    scan_kernel<<<(B_ * D_ / 2) / 8, 256, 0, stream>>>(
        delta, bcmat, gate_f, A_log, x, yf);

    // 7) layernorm -> bf16
    ln_kernel<<<M, 256, 0, stream>>>(yf, ln_g, ln_b, yb);

    // 8) out = y_ln @ W_out^T + b_out            (4096 x 1024 x 1024)
    gemm_big(yb, w_ot_b, b_out, out, D_, D_, 1);
}
